// EncoderBlock_14817637171306
// MI455X (gfx1250) — matrix operand order repaired and checked by reading, not compiled, not run on hardware
//
#include <hip/hip_runtime.h>
#include <math.h>

#define B_SZ  8
#define SEQ   2048
#define D_IN  300
#define H     128
#define H3    384
#define MROWS (B_SZ * SEQ)   // 16384
#define KPAD  320            // D_IN padded up to a multiple of 32
#define OUTC  640            // 5*H output channels

typedef __attribute__((ext_vector_type(16))) __bf16 v16bf;
typedef __attribute__((ext_vector_type(8)))  float  v8f;
typedef __attribute__((ext_vector_type(4)))  unsigned int u32x4;
typedef __attribute__((ext_vector_type(8)))  int i32x8;
typedef __attribute__((ext_vector_type(4)))  int i32x4;

// ---------------------------------------------------------------------------
// WMMA core: D = A(16x32 bf16) * B(32x16 bf16) + C(16x16 f32)
// ---------------------------------------------------------------------------
__device__ __forceinline__ v8f wmma_bf16(v16bf a, v16bf b, v8f c) {
  return __builtin_amdgcn_wmma_f32_16x16x32_bf16(
      /*neg_a=*/false, a, /*neg_b=*/false, b,
      /*c_mod=*/(short)0, c, /*reuse_a=*/false, /*reuse_b=*/false);
}

// A fragment (16x32, 16-bit): lane L holds row M=L&15; half=L>>4 selects the
// K-subblocks per ISA table (VGPR v<4: K=2v+8*half, v>=4: K=16+2(v-4)+8*half).
__device__ __forceinline__ v16bf load_frag_a(const __bf16* __restrict__ p, int ld) {
  const int lane = threadIdx.x & 31;
  const int m = lane & 15, half = lane >> 4;
  v16bf a;
#pragma unroll
  for (int v = 0; v < 8; ++v) {
    const int k = ((v >> 2) << 4) + (half << 3) + ((v & 3) << 1);
    a[2 * v]     = p[(size_t)m * ld + k];
    a[2 * v + 1] = p[(size_t)m * ld + k + 1];
  }
  return a;
}

// B fragment (32x16) from N-major storage Bt[n][k].  Lane: N=L&15;
// K per ISA table, same as A (v<4: K=2v+8*half+j, v>=4: K=16+2(v-4)+8*half+j).
__device__ __forceinline__ v16bf load_frag_b_nmajor(const __bf16* __restrict__ p, int ld) {
  const int lane = threadIdx.x & 31;
  const int n = lane & 15, half = lane >> 4;
  v16bf b;
#pragma unroll
  for (int v = 0; v < 8; ++v) {
    const int k = ((v >> 2) << 4) + (half << 3) + ((v & 3) << 1);
    b[2 * v]     = p[(size_t)n * ld + k];
    b[2 * v + 1] = p[(size_t)n * ld + k + 1];
  }
  return b;
}

// ---------------------------------------------------------------------------
// Tensor Data Mover: 2D tile load (tile_y rows x tile_x f32) -> LDS.
// Descriptor packed per CDNA5 ISA 8.3/8.4 (D# group0/group1), groups 2/3 zero
// (2D tensor).  data_size=4B; no padding/iterate/multicast.
// ---------------------------------------------------------------------------
__device__ __forceinline__ void tdm_load_2d_f32(unsigned int lds_off,
                                                const float* gptr,
                                                unsigned int tile_x,
                                                unsigned int tile_y,
                                                unsigned int row_stride_elems) {
  const unsigned long long ga = (unsigned long long)(uintptr_t)gptr;
  u32x4 g0;
  g0[0] = 1u;                                            // count=1 (valid user D#)
  g0[1] = lds_off;                                       // lds_addr (bytes)
  g0[2] = (unsigned int)(ga & 0xffffffffu);              // global_addr[31:0]
  g0[3] = (unsigned int)((ga >> 32) & 0x01ffffffu)       // global_addr[56:32]
        | (2u << 30);                                    // type=2 ("image")
  i32x8 g1;
  g1[0] = (int)(2u << 16);                               // data_size=2 (4 bytes)
  g1[1] = (int)((tile_x & 0xffffu) << 16);               // tensor_dim0[15:0]
  g1[2] = (int)((tile_y & 0xffffu) << 16);               // tensor_dim1[15:0] (dim0 hi=0)
  g1[3] = (int)((tile_x & 0xffffu) << 16);               // tile_dim0 (dim1 hi=0)
  g1[4] = (int)(tile_y & 0xffffu);                       // tile_dim1 (tile_dim2=0)
  g1[5] = (int)row_stride_elems;                         // tensor_dim0_stride[31:0]
  g1[6] = 0;                                             // stride0 hi / stride1 lo
  g1[7] = 0;                                             // stride1 hi
  const i32x4 z4 = {0, 0, 0, 0};
#if __clang_major__ >= 23
  const i32x8 z8 = {0, 0, 0, 0, 0, 0, 0, 0};
  __builtin_amdgcn_tensor_load_to_lds(g0, g1, z4, z4, z8, 0);
#else
  __builtin_amdgcn_tensor_load_to_lds(g0, g1, z4, z4, 0);
#endif
}

// ---------------------------------------------------------------------------
// fp32 -> bf16 convert with optional K padding (zeros)
// ---------------------------------------------------------------------------
__global__ void cvt_pad_kernel(const float* __restrict__ src, __bf16* __restrict__ dst,
                               int rows, int scol, int dcol) {
  const int idx = blockIdx.x * blockDim.x + threadIdx.x;
  if (idx >= rows * dcol) return;
  const int r = idx / dcol, c = idx - r * dcol;
  dst[idx] = (c < scol) ? (__bf16)src[(size_t)r * scol + c] : (__bf16)0.0f;
}

// ---------------------------------------------------------------------------
// P = seq @ W_lin^T + b_lin   (M=16384, N=128, K=320 padded)
// ---------------------------------------------------------------------------
__global__ void __launch_bounds__(256)
gemm_p_kernel(const __bf16* __restrict__ Abf, const __bf16* __restrict__ Bt,
              const float* __restrict__ bias, float* __restrict__ Cf,
              __bf16* __restrict__ Cbf) {
  const int wave = threadIdx.x >> 5, lane = threadIdx.x & 31;
  const int m0 = blockIdx.x * 128 + wave * 16;
  const int n = lane & 15, half = lane >> 4;
  for (int nt = 0; nt < 8; ++nt) {
    v8f acc = {0.f, 0.f, 0.f, 0.f, 0.f, 0.f, 0.f, 0.f};
    const __bf16* bp = Bt + (size_t)(nt * 16) * KPAD;
    const __bf16* ap = Abf + (size_t)m0 * KPAD;
    for (int kt = 0; kt < KPAD / 32; ++kt) {
      __builtin_prefetch(ap + (kt + 1) * 32, 0, 1);
      v16bf a   = load_frag_a(ap + kt * 32, KPAD);
      v16bf bfr = load_frag_b_nmajor(bp + kt * 32, KPAD);
      acc = wmma_bf16(a, bfr, acc);
    }
    const int col = nt * 16 + n;
    const float bb = bias[col];
#pragma unroll
    for (int v = 0; v < 8; ++v) {
      const int row = m0 + v + 8 * half;
      const float val = acc[v] + bb;
      Cf[(size_t)row * H + col]  = val;
      Cbf[(size_t)row * H + col] = (__bf16)val;
    }
  }
}

// ---------------------------------------------------------------------------
// x_proj = P @ Wih^T + bih, both directions (grid.z)  (M=16384, N=384, K=128)
// ---------------------------------------------------------------------------
__global__ void __launch_bounds__(256)
gemm_xproj_kernel(const __bf16* __restrict__ Pbf,
                  const __bf16* __restrict__ WihF, const __bf16* __restrict__ WihB,
                  const float* __restrict__ bihF, const float* __restrict__ bihB,
                  float* __restrict__ XF, float* __restrict__ XB) {
  const __bf16* Bt  = blockIdx.z ? WihB : WihF;
  const float* bias = blockIdx.z ? bihB : bihF;
  float* X          = blockIdx.z ? XB : XF;
  const int wave = threadIdx.x >> 5, lane = threadIdx.x & 31;
  const int m0 = blockIdx.x * 128 + wave * 16;
  const int nb = blockIdx.y * 128;
  const int n = lane & 15, half = lane >> 4;
  for (int nt = 0; nt < 8; ++nt) {
    v8f acc = {0.f, 0.f, 0.f, 0.f, 0.f, 0.f, 0.f, 0.f};
    const __bf16* bp = Bt + (size_t)(nb + nt * 16) * H;
    const __bf16* ap = Pbf + (size_t)m0 * H;
#pragma unroll
    for (int kt = 0; kt < H / 32; ++kt) {
      v16bf a   = load_frag_a(ap + kt * 32, H);
      v16bf bfr = load_frag_b_nmajor(bp + kt * 32, H);
      acc = wmma_bf16(a, bfr, acc);
    }
    const int col = nb + nt * 16 + n;
    const float bb = bias[col];
#pragma unroll
    for (int v = 0; v < 8; ++v) {
      const int row = m0 + v + 8 * half;
      X[(size_t)row * H3 + col] = acc[v] + bb;
    }
  }
}

// ---------------------------------------------------------------------------
// Sequential GRU scan, one workgroup per direction (blockIdx.x = dir).
// h (8x128, M padded to 16) lives in LDS; Whh stays L2-resident (bf16).
// The next timestep's x_proj tile (8 rows x 384 f32, row stride SEQ*H3) is
// DMA'd into a double-buffered LDS staging area by the Tensor Data Mover one
// step ahead, synchronized with s_wait_tensorcnt — this takes the global-load
// latency off the 2048-step sequential critical path.
// Writes R directly into out[:, :, dir*128 : dir*128+128].
// ---------------------------------------------------------------------------
__global__ void __launch_bounds__(256)
gru_scan_kernel(const float* __restrict__ XF, const float* __restrict__ XB,
                const __bf16* __restrict__ WhhF, const __bf16* __restrict__ WhhB,
                const float* __restrict__ bhhF, const float* __restrict__ bhhB,
                float* __restrict__ out) {
  const int dir = blockIdx.x;
  const float* Xp    = dir ? XB : XF;
  const __bf16* Whh  = dir ? WhhB : WhhF;
  const float* bhh   = dir ? bhhB : bhhF;

  __shared__ __bf16 h_bf[16 * H];       // matrix feed (rows 8..15 stay zero)
  __shared__ float  gh[16 * H3];        // h @ Whh^T result tile
  __shared__ float  h_f[8 * H];         // fp32 master copy of hidden state
  __shared__ float  xbuf[2][8 * H3];    // TDM double buffer for x_proj[t]

  const int tid = threadIdx.x;
  for (int i = tid; i < 16 * H; i += 256) h_bf[i] = (__bf16)0.0f;
  for (int i = tid; i < 8 * H; i += 256)  h_f[i] = 0.0f;
  __syncthreads();

  const int wave = tid >> 5, lane = tid & 31;
  const int n = lane & 15, half = lane >> 4;

  // Prologue: wave 0 kicks off the DMA for timestep 0 (TDM ignores EXEC; one
  // issue per wave, so restrict to wave 0 to avoid duplicate transfers).
  if (tid < 32) {
    const int s0 = dir ? (SEQ - 1) : 0;
    tdm_load_2d_f32((unsigned int)(uintptr_t)&xbuf[0][0],
                    Xp + (size_t)s0 * H3, H3, 8, SEQ * H3);
  }

  for (int t = 0; t < SEQ; ++t) {
    const int s = dir ? (SEQ - 1 - t) : t;
    const int buf = t & 1;

    // ---- recurrent matmul: gh = h @ Whh^T (96 WMMAs across 8 waves)
    v16bf afr[4];
#pragma unroll
    for (int kt = 0; kt < 4; ++kt) afr[kt] = load_frag_a(h_bf + kt * 32, H);

#pragma unroll
    for (int i = 0; i < 3; ++i) {
      const int n0 = (wave + 8 * i) * 16;
      v8f acc = {0.f, 0.f, 0.f, 0.f, 0.f, 0.f, 0.f, 0.f};
#pragma unroll
      for (int kt = 0; kt < 4; ++kt) {
        v16bf bfr = load_frag_b_nmajor(Whh + (size_t)n0 * H + kt * 32, H);
        acc = wmma_bf16(afr[kt], bfr, acc);
      }
#pragma unroll
      for (int v = 0; v < 8; ++v)
        gh[(v + 8 * half) * H3 + n0 + n] = acc[v];
    }

    // ---- wait for X[t] DMA (no-op for waves with TENSORcnt==0), publish gh
    __builtin_amdgcn_s_wait_tensorcnt(0);
    __syncthreads();

    // ---- issue DMA for X[t+1] into the other buffer while gates compute
    if (t + 1 < SEQ && tid < 32) {
      const int sn = dir ? (SEQ - 2 - t) : (t + 1);
      tdm_load_2d_f32((unsigned int)(uintptr_t)&xbuf[buf ^ 1][0],
                      Xp + (size_t)sn * H3, H3, 8, SEQ * H3);
    }

    // ---- elementwise GRU gates on 8x128 state
    const float* xb = &xbuf[buf][0];
    for (int i = tid; i < 8 * H; i += 256) {
      const int m = i >> 7, c = i & 127;
      const float xr = xb[m * H3 + c];
      const float xz = xb[m * H3 + H + c];
      const float xn = xb[m * H3 + 2 * H + c];
      const float hr = gh[m * H3 + c]         + bhh[c];
      const float hz = gh[m * H3 + H + c]     + bhh[H + c];
      const float hn = gh[m * H3 + 2 * H + c] + bhh[2 * H + c];
      const float r  = 1.0f / (1.0f + expf(-(xr + hr)));
      const float z  = 1.0f / (1.0f + expf(-(xz + hz)));
      const float nn = tanhf(xn + r * hn);
      const float hv = (1.0f - z) * nn + z * h_f[i];
      h_f[i] = hv;
      h_bf[m * H + c] = (__bf16)hv;
      out[((size_t)m * SEQ + s) * OUTC + dir * H + c] = hv;
    }
    __syncthreads();
  }
}

// ---------------------------------------------------------------------------
// Assemble G = [P | Rf | Rb] as bf16 (row-major AND transposed), Gw = G * w_f,
// A = G.w_a, B = G.w_b.  One wave per row; wave32 shfl reduction.
// ---------------------------------------------------------------------------
__global__ void __launch_bounds__(256)
build_g_kernel(const float* __restrict__ Pf, const float* __restrict__ out,
               const float* __restrict__ wa, const float* __restrict__ wb,
               const float* __restrict__ wf,
               __bf16* __restrict__ Gbf, __bf16* __restrict__ Gwbf,
               __bf16* __restrict__ Gt,
               float* __restrict__ Ab, float* __restrict__ Bb) {
  const int wave = threadIdx.x >> 5, lane = threadIdx.x & 31;
  const int row = blockIdx.x * 8 + wave;
  const int b = row >> 11, jj = row & (SEQ - 1);
  float asum = 0.f, bsum = 0.f;
  for (int i = lane; i < H3; i += 32) {
    const float g = (i < H) ? Pf[(size_t)row * H + i]
                            : out[(size_t)row * OUTC + (i - H)];
    const __bf16 gb = (__bf16)g;
    Gbf[(size_t)row * H3 + i]  = gb;
    Gwbf[(size_t)row * H3 + i] = (__bf16)(g * wf[i]);
    Gt[((size_t)b * H3 + i) * SEQ + jj] = gb;   // transposed copy for Q-GEMM
    asum += g * wa[i];
    bsum += g * wb[i];
  }
#pragma unroll
  for (int off = 16; off > 0; off >>= 1) {
    asum += __shfl_down(asum, off);
    bsum += __shfl_down(bsum, off);
  }
  if (lane == 0) { Ab[row] = asum; Bb[row] = bsum; }
}

// ---------------------------------------------------------------------------
// S[b,j,k] = A[b,j] + B[b,k] + Gw[b,j,:] . G[b,k,:]   (2048x2048x384 per batch)
// ---------------------------------------------------------------------------
__global__ void __launch_bounds__(256)
gemm_s_kernel(const __bf16* __restrict__ Gwbf, const __bf16* __restrict__ Gbf,
              const float* __restrict__ Ab, const float* __restrict__ Bb,
              float* __restrict__ Sbuf) {
  const int b = blockIdx.z;
  const __bf16* Ap = Gwbf + (size_t)b * SEQ * H3;
  const __bf16* Bp = Gbf  + (size_t)b * SEQ * H3;
  const float* Abias = Ab + (size_t)b * SEQ;
  const float* Bbias = Bb + (size_t)b * SEQ;
  float* Sb = Sbuf + (size_t)b * SEQ * SEQ;
  const int wave = threadIdx.x >> 5, lane = threadIdx.x & 31;
  const int j0 = blockIdx.x * 128 + wave * 16;
  const int k0b = blockIdx.y * 128;
  const int n = lane & 15, half = lane >> 4;
  for (int nt = 0; nt < 8; ++nt) {
    v8f acc = {0.f, 0.f, 0.f, 0.f, 0.f, 0.f, 0.f, 0.f};
    const __bf16* bp = Bp + (size_t)(k0b + nt * 16) * H3;
    const __bf16* ap = Ap + (size_t)j0 * H3;
#pragma unroll 1
    for (int kt = 0; kt < H3 / 32; ++kt) {
      __builtin_prefetch(ap + (kt + 1) * 32, 0, 1);
      v16bf a   = load_frag_a(ap + kt * 32, H3);
      v16bf bfr = load_frag_b_nmajor(bp + kt * 32, H3);
      acc = wmma_bf16(a, bfr, acc);
    }
    const int kcol = k0b + nt * 16 + n;
    const float bbias = Bbias[kcol];
#pragma unroll
    for (int v = 0; v < 8; ++v) {
      const int row = j0 + v + 8 * half;
      Sb[(size_t)row * SEQ + kcol] = acc[v] + Abias[row] + bbias;
    }
  }
}

// ---------------------------------------------------------------------------
// Column-wise softmax (torch Softmax(dim=1)): one column per thread, fully
// coalesced across adjacent k; emits bf16 probabilities for the next WMMA.
// ---------------------------------------------------------------------------
__global__ void __launch_bounds__(256)
softmax_col_kernel(const float* __restrict__ Sbuf, __bf16* __restrict__ Att) {
  const int b = blockIdx.y;
  const int k = blockIdx.x * blockDim.x + threadIdx.x;
  const float* Sb = Sbuf + (size_t)b * SEQ * SEQ;
  __bf16* Ab = Att + (size_t)b * SEQ * SEQ;
  float m = -3.4e38f;
  for (int j = 0; j < SEQ; ++j) m = fmaxf(m, Sb[(size_t)j * SEQ + k]);
  float sum = 0.f;
  for (int j = 0; j < SEQ; ++j) sum += expf(Sb[(size_t)j * SEQ + k] - m);
  const float inv = 1.0f / sum;
  for (int j = 0; j < SEQ; ++j)
    Ab[(size_t)j * SEQ + k] = (__bf16)(expf(Sb[(size_t)j * SEQ + k] - m) * inv);
}

// ---------------------------------------------------------------------------
// Q = Att @ G  (M=2048, N=384, K=2048 per batch); B operand comes from the
// transposed copy Gt so fragments load as contiguous b128s.  Writes
// out[:, :, 256:640].
// ---------------------------------------------------------------------------
__global__ void __launch_bounds__(256)
gemm_q_kernel(const __bf16* __restrict__ Att, const __bf16* __restrict__ Gt,
              float* __restrict__ out) {
  const int b = blockIdx.z;
  const __bf16* Ap  = Att + (size_t)b * SEQ * SEQ;
  const __bf16* Gtb = Gt  + (size_t)b * H3 * SEQ;
  const int wave = threadIdx.x >> 5, lane = threadIdx.x & 31;
  const int j0 = blockIdx.x * 128 + wave * 16;
  const int d0 = blockIdx.y * 128;
  const int n = lane & 15, half = lane >> 4;
  for (int nt = 0; nt < 8; ++nt) {
    const int dcol0 = d0 + nt * 16;
    v8f acc = {0.f, 0.f, 0.f, 0.f, 0.f, 0.f, 0.f, 0.f};
    const __bf16* ap = Ap + (size_t)j0 * SEQ;
    const __bf16* bp = Gtb + (size_t)dcol0 * SEQ;
#pragma unroll 1
    for (int kt = 0; kt < SEQ / 32; ++kt) {
      __builtin_prefetch(ap + (kt + 1) * 32, 0, 1);
      v16bf a   = load_frag_a(ap + kt * 32, SEQ);
      v16bf bfr = load_frag_b_nmajor(bp + kt * 32, SEQ);
      acc = wmma_bf16(a, bfr, acc);
    }
    const int d = dcol0 + n;
#pragma unroll
    for (int v = 0; v < 8; ++v) {
      const int row = j0 + v + 8 * half;
      out[((size_t)b * SEQ + row) * OUTC + 2 * H + d] = acc[v];
    }
  }
}

// ---------------------------------------------------------------------------
extern "C" void kernel_launch(void* const* d_in, const int* in_sizes, int n_in,
                              void* d_out, int out_size, void* d_ws, size_t ws_size,
                              hipStream_t stream) {
  (void)in_sizes; (void)n_in; (void)out_size; (void)ws_size;
  const float* seq   = (const float*)d_in[0];
  const float* W_lin = (const float*)d_in[1];
  const float* b_lin = (const float*)d_in[2];
  const float* Wih_f = (const float*)d_in[3];
  const float* Whh_f = (const float*)d_in[4];
  const float* bih_f = (const float*)d_in[5];
  const float* bhh_f = (const float*)d_in[6];
  const float* Wih_b = (const float*)d_in[7];
  const float* Whh_b = (const float*)d_in[8];
  const float* bih_b = (const float*)d_in[9];
  const float* bhh_b = (const float*)d_in[10];
  const float* w_a   = (const float*)d_in[11];
  const float* w_b   = (const float*)d_in[12];
  const float* w_f   = (const float*)d_in[13];
  float* out = (float*)d_out;
  char* ws = (char*)d_ws;

  // ---- workspace layout (early region is dead by softmax time; Att aliases it)
  size_t off = 0;
  auto take = [&](size_t bytes) -> size_t {
    size_t o = off; off = (off + bytes + 255) & ~(size_t)255; return o;
  };
  const size_t o_seqbf = take((size_t)MROWS * KPAD * 2);   // 10.5 MB
  const size_t o_wlin  = take((size_t)H * KPAD * 2);
  const size_t o_pbf   = take((size_t)MROWS * H * 2);      // 4.2 MB
  const size_t o_pf    = take((size_t)MROWS * H * 4);      // 8.4 MB
  const size_t o_wihf  = take((size_t)H3 * H * 2);
  const size_t o_wihb  = take((size_t)H3 * H * 2);
  const size_t o_whhf  = take((size_t)H3 * H * 2);
  const size_t o_whhb  = take((size_t)H3 * H * 2);
  const size_t o_xpf   = take((size_t)MROWS * H3 * 4);     // 25.2 MB
  const size_t o_xpb   = take((size_t)MROWS * H3 * 4);     // 25.2 MB
  // persistent region
  const size_t o_gbf   = take((size_t)MROWS * H3 * 2);
  const size_t o_gwbf  = take((size_t)MROWS * H3 * 2);
  const size_t o_gt    = take((size_t)MROWS * H3 * 2);
  const size_t o_ab    = take((size_t)MROWS * 4);
  const size_t o_bb    = take((size_t)MROWS * 4);
  const size_t o_s     = take((size_t)B_SZ * SEQ * SEQ * 4); // 134 MB
  const size_t o_att   = 0; // bf16 Att (67 MB) aliases the dead early region

  __bf16* seqbf  = (__bf16*)(ws + o_seqbf);
  __bf16* wlinbf = (__bf16*)(ws + o_wlin);
  __bf16* pbf    = (__bf16*)(ws + o_pbf);
  float*  pf     = (float*)(ws + o_pf);
  __bf16* wihfbf = (__bf16*)(ws + o_wihf);
  __bf16* wihbbf = (__bf16*)(ws + o_wihb);
  __bf16* whhfbf = (__bf16*)(ws + o_whhf);
  __bf16* whhbbf = (__bf16*)(ws + o_whhb);
  float*  xpf    = (float*)(ws + o_xpf);
  float*  xpb    = (float*)(ws + o_xpb);
  __bf16* gbf    = (__bf16*)(ws + o_gbf);
  __bf16* gwbf   = (__bf16*)(ws + o_gwbf);
  __bf16* gt     = (__bf16*)(ws + o_gt);
  float*  ab     = (float*)(ws + o_ab);
  float*  bb     = (float*)(ws + o_bb);
  float*  sbuf   = (float*)(ws + o_s);
  __bf16* att    = (__bf16*)(ws + o_att);

  const dim3 blk(256);

  // 0) bf16 conversions (K-padded where needed)
  cvt_pad_kernel<<<dim3(((unsigned)(MROWS * KPAD) + 255) / 256), blk, 0, stream>>>(
      seq, seqbf, MROWS, D_IN, KPAD);
  cvt_pad_kernel<<<dim3((H * KPAD + 255) / 256), blk, 0, stream>>>(
      W_lin, wlinbf, H, D_IN, KPAD);
  cvt_pad_kernel<<<dim3((H3 * H + 255) / 256), blk, 0, stream>>>(Wih_f, wihfbf, H3, H, H);
  cvt_pad_kernel<<<dim3((H3 * H + 255) / 256), blk, 0, stream>>>(Wih_b, wihbbf, H3, H, H);
  cvt_pad_kernel<<<dim3((H3 * H + 255) / 256), blk, 0, stream>>>(Whh_f, whhfbf, H3, H, H);
  cvt_pad_kernel<<<dim3((H3 * H + 255) / 256), blk, 0, stream>>>(Whh_b, whhbbf, H3, H, H);

  // 1) P = seq @ W_lin^T + b_lin
  gemm_p_kernel<<<dim3(MROWS / 128), blk, 0, stream>>>(seqbf, wlinbf, b_lin, pf, pbf);

  // 2) x_proj (both directions)
  gemm_xproj_kernel<<<dim3(MROWS / 128, H3 / 128, 2), blk, 0, stream>>>(
      pbf, wihfbf, wihbbf, bih_f, bih_b, xpf, xpb);

  // 3) sequential GRU scans (TDM-staged X, LDS-resident state) -> out[:, :, 0:256]
  gru_scan_kernel<<<dim3(2), blk, 0, stream>>>(xpf, xpb, whhfbf, whhbbf, bhh_f, bhh_b, out);

  // 4) build G/Gw/Gt (bf16) and the A/B bias vectors
  build_g_kernel<<<dim3(MROWS / 8), blk, 0, stream>>>(
      pf, out, w_a, w_b, w_f, gbf, gwbf, gt, ab, bb);

  // 5) S = A[j] + B[k] + Gw G^T
  gemm_s_kernel<<<dim3(SEQ / 128, SEQ / 128, B_SZ), blk, 0, stream>>>(
      gwbf, gbf, ab, bb, sbuf);

  // 6) column softmax -> bf16 Att
  softmax_col_kernel<<<dim3(SEQ / 256, B_SZ), blk, 0, stream>>>(sbuf, att);

  // 7) Q = Att @ G (transposed B operand) -> out[:, :, 256:640]
  gemm_q_kernel<<<dim3(SEQ / 128, H3 / 128, B_SZ), blk, 0, stream>>>(att, gt, out);
}
